// MultistepModel_35270271435151
// MI455X (gfx1250) — matrix operand order repaired and checked by reading, not compiled, not run on hardware
//
#include <hip/hip_runtime.h>
#include <stdint.h>

#define V_    50257
#define H_    256
#define M_    512
#define B_    16
#define T_    4096
#define NPADV 50272           // V padded to multiple of 16
#define NSC   (T_ - 3)        // scored positions per batch

typedef __attribute__((ext_vector_type(16))) __bf16 v16bf;
typedef __attribute__((ext_vector_type(8)))  float  v8f;

static __device__ __forceinline__ v8f wmma_bf16(v16bf a, v16bf b, v8f c) {
  // D = A(16x32 bf16) * B(32x16 bf16) + C(16x16 f32)
  return __builtin_amdgcn_wmma_f32_16x16x32_bf16(false, a, false, b, (short)0, c,
                                                 false, false);
}

// ---------------------------------------------------------------------------
// Pre-swizzle an fp32 row-major (K x N) weight into bf16 B-fragment order:
// fragment f = kblk*(Npad/16)+nblk ; within fragment: lane 0..31, elem i 0..15
//   K = kblk*32 + (lane&16 ? 8 : 0) + (i&7) + (i&8 ? 16 : 0) ;  Ncol = nblk*16 + (lane&15)
// ---------------------------------------------------------------------------
__global__ void ms_swizzle_kernel(const float* __restrict__ W, __bf16* __restrict__ outw,
                                  int K, int N, int Npad) {
  size_t total = (size_t)K * Npad;
  int nt = Npad >> 4;
  for (size_t idx = (size_t)blockIdx.x * blockDim.x + threadIdx.x; idx < total;
       idx += (size_t)gridDim.x * blockDim.x) {
    size_t f  = idx >> 9;
    int lane  = (int)((idx >> 4) & 31);
    int i     = (int)(idx & 15);
    int kblk  = (int)(f / nt);
    int nblk  = (int)(f % nt);
    int k     = kblk * 32 + ((lane & 16) ? 8 : 0) + (i & 7) + ((i & 8) ? 16 : 0);
    int n     = nblk * 16 + (lane & 15);
    if (n >= N) n = N - 1;                      // pad columns (stores are guarded later)
    outw[idx] = (__bf16)W[(size_t)k * N + n];
  }
}

// ---------------------------------------------------------------------------
// Fused: async embed gather -> FFN (WMMA bf16, 2 token-tiles per wave for
// 2x B-fragment reuse; tile0 A-frags in VGPRs, tile1 A-frags served from LDS
// in fragment-major order) -> residual -> LN -> gate score.
// LDS caps occupancy anyway, so waves_per_eu(1) frees the full VGPR budget
// (kills the scratch spill seen in round 3).
// ---------------------------------------------------------------------------
__global__ void __attribute__((amdgpu_waves_per_eu(1)))
ms_ffn_ln_kernel(const int* __restrict__ seq, const float* __restrict__ embed,
                 const __bf16* __restrict__ w1b, const float* __restrict__ b1,
                 const __bf16* __restrict__ w2b, const float* __restrict__ b2,
                 const float* __restrict__ ln_g, const float* __restrict__ ln_b,
                 const float* __restrict__ gate_w, const float* __restrict__ gate_b,
                 float* __restrict__ hidden, float* __restrict__ scores) {
  __shared__ float  hx[2][16][260];        // h tiles fp32 (become x = h + ffn)
  __shared__ __bf16 y1f[2][16][32][16];    // relu(h@W1+b1), A-fragment-major
  __shared__ __bf16 habuf1[8][32][16];     // tile1 h A-fragments
  __shared__ float  smu[2][16], srstd[2][16];

  const int lane   = threadIdx.x;
  const int token0 = blockIdx.x * 32;

  // ---- async gather of 32 embedding rows straight into LDS (ASYNCcnt) ----
  {
    int r = lane >> 1, half = lane & 1;
#pragma unroll
    for (int t = 0; t < 2; ++t) {
      int token = token0 + t * 16 + r;
      int id = seq[token];
      const float* src = embed + (size_t)id * H_ + half * 128;
      unsigned ldst = (unsigned)(uintptr_t)(const void*)&hx[t][r][half * 128];
#pragma unroll
      for (int i = 0; i < 32; ++i) {
        asm volatile("global_load_async_to_lds_b128 %0, %1, off"
                     :: "v"(ldst + i * 16), "v"(src + i * 4) : "memory");
      }
    }
  }
  asm volatile("s_wait_asynccnt 0x0" ::: "memory");
  __syncthreads();

  const int row  = lane & 15;
  const int coff = (lane & 16) ? 8 : 0;   // A-frag K sub-offset
  const int col  = lane & 15;             // D col
  const int roff = (lane & 16) ? 8 : 0;   // D row offset

  // ---- tile0 A-frags -> registers; tile1 A-frags -> LDS fragment-major ----
  v16bf afr0[8];
#pragma unroll
  for (int kk = 0; kk < 8; ++kk) {
    const float* p0 = &hx[0][row][kk * 32 + coff];
    const float* p1 = &hx[1][row][kk * 32 + coff];
    v16bf a0, a1;
#pragma unroll
    for (int i = 0; i < 8; ++i) {
      a0[i] = (__bf16)p0[i];  a0[i + 8] = (__bf16)p0[16 + i];
      a1[i] = (__bf16)p1[i];  a1[i + 8] = (__bf16)p1[16 + i];
    }
    afr0[kk] = a0;
    *(v16bf*)&habuf1[kk][lane][0] = a1;    // same-lane produce/consume
  }
  __syncthreads();

  // ---- FFN1: y1 = relu(h @ W1 + b1); each B-fragment feeds 2 WMMAs ----
  for (int n = 0; n < 32; ++n) {
    unsigned lidx = 0;
    asm volatile("" : "+v"(lidx));         // unknown index: blocks LICM re-hoist,
                                           // but keeps LDS provenance (ds_load path)
    const v16bf* h1p = (const v16bf*)&habuf1[0][lane][0] + lidx;
    v8f acc0 = {}, acc1 = {};
#pragma unroll
    for (int kk = 0; kk < 8; ++kk) {
      v16bf bf = *(const v16bf*)(w1b + ((size_t)(kk * 32 + n) * 32 + lane) * 16);
      if (kk < 7) __builtin_prefetch(w1b + ((size_t)((kk + 1) * 32 + n) * 32 + lane) * 16, 0, 3);
      acc0 = wmma_bf16(afr0[kk], bf, acc0);
      acc1 = wmma_bf16(h1p[kk * 32], bf, acc1);
    }
    // epilogue: bias + relu, scatter D-layout -> A-fragment layout of y1f
    float bias = b1[n * 16 + col];
    int e   = (col & 7) + ((n & 1) ? 8 : 0);
    int l2b = (col >= 8) ? 16 : 0;
    int kk2 = n >> 1;
#pragma unroll
    for (int j = 0; j < 8; ++j) {
      float v0 = acc0[j] + bias, v1 = acc1[j] + bias;
      y1f[0][kk2][l2b + roff + j][e] = (__bf16)(v0 > 0.f ? v0 : 0.f);
      y1f[1][kk2][l2b + roff + j][e] = (__bf16)(v1 > 0.f ? v1 : 0.f);
    }
  }
  __syncthreads();

  // ---- FFN2: x = h + (y1 @ W2 + b2); A-frags are single 32B LDS loads ----
  for (int n = 0; n < 16; ++n) {
    unsigned lidx = 0;
    asm volatile("" : "+v"(lidx));
    const v16bf* y0p = (const v16bf*)&y1f[0][0][lane][0] + lidx;
    const v16bf* y1p = (const v16bf*)&y1f[1][0][lane][0] + lidx;
    v8f acc0 = {}, acc1 = {};
#pragma unroll
    for (int kk = 0; kk < 16; ++kk) {
      v16bf bf = *(const v16bf*)(w2b + ((size_t)(kk * 16 + n) * 32 + lane) * 16);
      acc0 = wmma_bf16(y0p[kk * 32], bf, acc0);
      acc1 = wmma_bf16(y1p[kk * 32], bf, acc1);
    }
    float bias = b2[n * 16 + col];
#pragma unroll
    for (int j = 0; j < 8; ++j) {
      hx[0][roff + j][n * 16 + col] += acc0[j] + bias;
      hx[1][roff + j][n * 16 + col] += acc1[j] + bias;
    }
  }
  __syncthreads();

  // ---- LayerNorm stats: 2 lanes per row ----
#pragma unroll
  for (int t = 0; t < 2; ++t) {
    int r = lane & 15, half = lane >> 4;
    float sum = 0.f, ss = 0.f;
    const float* p = &hx[t][r][half * 128];
    for (int c = 0; c < 128; ++c) { float v = p[c]; sum += v; ss += v * v; }
    sum += __shfl_xor(sum, 16, 32);
    ss  += __shfl_xor(ss, 16, 32);
    float mu  = sum * (1.f / H_);
    float var = ss * (1.f / H_) - mu * mu;
    if (lane < 16) { smu[t][r] = mu; srstd[t][r] = rsqrtf(var + 1e-5f); }
  }
  __syncthreads();

  // ---- normalize, write hidden, gate score (coalesced per row) ----
  const float gb = gate_b[0];
  for (int t = 0; t < 2; ++t) {
    for (int r = 0; r < 16; ++r) {
      int token = token0 + t * 16 + r;
      float mu = smu[t][r], rs = srstd[t][r];
      float sc = 0.f;
      for (int c = lane; c < H_; c += 32) {
        float v = (hx[t][r][c] - mu) * rs * ln_g[c] + ln_b[c];
        hidden[(size_t)token * H_ + c] = v;
        sc += v * gate_w[c];
      }
#pragma unroll
      for (int off = 16; off >= 1; off >>= 1) sc += __shfl_xor(sc, off, 32);
      if (lane == 0) {
        int pos = token & (T_ - 1);
        scores[token] = (pos < T_ - 3) ? (sc + gb) : -1.0e30f;
      }
    }
  }
}

// ---------------------------------------------------------------------------
// Per-batch top-M selection via value bisection (softmax is permutation-
// invariant, so only the selected SET matters -- no sort needed).
// ---------------------------------------------------------------------------
__global__ void ms_topk_kernel(const float* __restrict__ scores, int* __restrict__ topidx) {
  const int b = blockIdx.x, tid = threadIdx.x;
  __shared__ float s[NSC];
  __shared__ float redf[256];
  __shared__ int   redi[256];
  __shared__ int   scnt;

  const float* sb = scores + (size_t)b * T_;
  for (int i = tid; i < NSC; i += 256) s[i] = sb[i];
  __syncthreads();

  float mx = -1e30f, mn = 1e30f;
  for (int i = tid; i < NSC; i += 256) { float v = s[i]; mx = fmaxf(mx, v); mn = fminf(mn, v); }
  redf[tid] = mx; __syncthreads();
  for (int o = 128; o; o >>= 1) { if (tid < o) redf[tid] = fmaxf(redf[tid], redf[tid + o]); __syncthreads(); }
  float hi = redf[0]; __syncthreads();
  redf[tid] = mn; __syncthreads();
  for (int o = 128; o; o >>= 1) { if (tid < o) redf[tid] = fminf(redf[tid], redf[tid + o]); __syncthreads(); }
  float lo = redf[0] - 1.f; __syncthreads();

  for (int it = 0; it < 64; ++it) {            // invariant: cnt(>lo)>=M, cnt(>hi)<M
    float mid = 0.5f * (lo + hi);
    int c = 0;
    for (int i = tid; i < NSC; i += 256) c += (s[i] > mid);
    redi[tid] = c; __syncthreads();
    for (int o = 128; o; o >>= 1) { if (tid < o) redi[tid] += redi[tid + o]; __syncthreads(); }
    int cnt = redi[0]; __syncthreads();
    if (cnt >= M_) lo = mid; else hi = mid;
  }

  if (tid == 0) scnt = 0;
  __syncthreads();
  int* ob = topidx + b * M_;
  for (int i = tid; i < NSC; i += 256)
    if (s[i] > hi) { int k = atomicAdd(&scnt, 1); if (k < M_) ob[k] = i; }
  __syncthreads();
  for (int i = tid; i < NSC; i += 256)
    if (s[i] <= hi && s[i] > lo) { int k = atomicAdd(&scnt, 1); if (k < M_) ob[k] = i; }
  __syncthreads();
  for (int k = scnt + tid; k < M_; k += 256) ob[k] = 0;   // safety fill
}

// ---------------------------------------------------------------------------
// q = (hidden[b,T-2] @ qa_w + qa_b) @ qr_w + qr_b
// ---------------------------------------------------------------------------
__global__ void ms_query_kernel(const float* __restrict__ hidden,
                                const float* __restrict__ qa_w, const float* __restrict__ qa_b,
                                const float* __restrict__ qr_w, const float* __restrict__ qr_b,
                                float* __restrict__ qbuf) {
  const int b = blockIdx.x, tid = threadIdx.x;
  __shared__ float hrow[H_], qh[H_];
  hrow[tid] = hidden[((size_t)b * T_ + (T_ - 2)) * H_ + tid];
  __syncthreads();
  float a = qa_b[tid];
  for (int k = 0; k < H_; ++k) a += hrow[k] * qa_w[(size_t)k * H_ + tid];
  qh[tid] = a;
  __syncthreads();
  float a2 = qr_b[tid];
  for (int k = 0; k < H_; ++k) a2 += qh[k] * qr_w[(size_t)k * H_ + tid];
  qbuf[b * H_ + tid] = a2;
}

// ---------------------------------------------------------------------------
// att = memory.q ; softmax ; ctx = attn^T memory  -> ctx as bf16 (16x256)
// ---------------------------------------------------------------------------
__global__ void ms_attn_kernel(const float* __restrict__ hidden, const int* __restrict__ topidx,
                               const float* __restrict__ qbuf, __bf16* __restrict__ ctxb) {
  const int b = blockIdx.x, tid = threadIdx.x;
  __shared__ int   srows[M_];
  __shared__ float sq[H_];
  __shared__ float satt[M_];
  __shared__ float red[256];

  sq[tid]          = qbuf[b * H_ + tid];
  srows[tid]       = topidx[b * M_ + tid];
  srows[tid + 256] = topidx[b * M_ + 256 + tid];
  __syncthreads();

  for (int m = tid; m < M_; m += 256) {
    const float* mem = hidden + ((size_t)b * T_ + srows[m]) * H_;
    float a = 0.f;
    for (int c = 0; c < H_; ++c) a += mem[c] * sq[c];
    satt[m] = a;
  }
  __syncthreads();

  float mx = fmaxf(satt[tid], satt[tid + 256]);
  red[tid] = mx; __syncthreads();
  for (int o = 128; o; o >>= 1) { if (tid < o) red[tid] = fmaxf(red[tid], red[tid + o]); __syncthreads(); }
  mx = red[0]; __syncthreads();

  float e0 = __expf(satt[tid] - mx), e1 = __expf(satt[tid + 256] - mx);
  satt[tid] = e0; satt[tid + 256] = e1;
  red[tid] = e0 + e1; __syncthreads();
  for (int o = 128; o; o >>= 1) { if (tid < o) red[tid] += red[tid + o]; __syncthreads(); }
  float inv = 1.f / red[0];
  __syncthreads();

  float cv = 0.f;
  for (int m = 0; m < M_; ++m)
    cv += satt[m] * hidden[((size_t)b * T_ + srows[m]) * H_ + tid];   // coalesced in tid
  ctxb[b * H_ + tid] = (__bf16)(cv * inv);
}

// ---------------------------------------------------------------------------
// out = ctx(16x256) @ out_w(256xV) + out_b  -- WMMA, one wave per 16-col tile
// ---------------------------------------------------------------------------
__global__ void ms_outproj_kernel(const __bf16* __restrict__ ctxb, const __bf16* __restrict__ owb,
                                  const float* __restrict__ out_b, float* __restrict__ out) {
  const int lane  = threadIdx.x & 31;
  const int ntile = blockIdx.x * 4 + (threadIdx.x >> 5);
  const int NT    = NPADV / 16;
  if (ntile >= NT) return;                      // wave-uniform

  const int row  = lane & 15;
  const int coff = (lane & 16) ? 8 : 0;
  v16bf afr[8];
#pragma unroll
  for (int kk = 0; kk < 8; ++kk) {
    const __bf16* p = ctxb + row * H_ + kk * 32 + coff;
#pragma unroll
    for (int i = 0; i < 8; ++i) { afr[kk][i] = p[i]; afr[kk][i + 8] = p[16 + i]; }
  }
  v8f acc = {};
#pragma unroll
  for (int kk = 0; kk < 8; ++kk) {
    const v16bf* bp = (const v16bf*)(owb + ((size_t)(kk * NT + ntile) * 32 + lane) * 16);
    acc = wmma_bf16(afr[kk], *bp, acc);
  }
  const int col = lane & 15;
  const int v   = ntile * 16 + col;
  const int roff = (lane & 16) ? 8 : 0;
  if (v < V_) {
    float ob = out_b[v];
#pragma unroll
    for (int j = 0; j < 8; ++j) out[(size_t)(roff + j) * V_ + v] = acc[j] + ob;
  }
}

// ---------------------------------------------------------------------------
extern "C" void kernel_launch(void* const* d_in, const int* in_sizes, int n_in,
                              void* d_out, int out_size, void* d_ws, size_t ws_size,
                              hipStream_t stream) {
  const int*   seq    = (const int*)  d_in[0];
  const float* embed  = (const float*)d_in[1];
  const float* W1     = (const float*)d_in[2];
  const float* b1     = (const float*)d_in[3];
  const float* W2     = (const float*)d_in[4];
  const float* b2     = (const float*)d_in[5];
  const float* ln_g   = (const float*)d_in[6];
  const float* ln_b   = (const float*)d_in[7];
  const float* gate_w = (const float*)d_in[8];
  const float* gate_b = (const float*)d_in[9];
  const float* qa_w   = (const float*)d_in[10];
  const float* qa_b   = (const float*)d_in[11];
  const float* qr_w   = (const float*)d_in[12];
  const float* qr_b   = (const float*)d_in[13];
  const float* out_w  = (const float*)d_in[14];
  const float* out_b  = (const float*)d_in[15];
  float* out = (float*)d_out;

  // ---- workspace layout (all chunks 32B-aligned) ----
  uint8_t* w = (uint8_t*)d_ws;
  float* hidden = (float*)w;  w += (size_t)B_ * T_ * H_ * sizeof(float);   // 64 MB
  float* scores = (float*)w;  w += (size_t)B_ * T_ * sizeof(float);
  int*   topidx = (int*)w;    w += (size_t)B_ * M_ * sizeof(int);
  float* qbuf   = (float*)w;  w += (size_t)B_ * H_ * sizeof(float);
  __bf16* ctxb  = (__bf16*)w; w += (size_t)B_ * H_ * sizeof(__bf16);
  w = (uint8_t*)d_ws + (((size_t)(w - (uint8_t*)d_ws) + 255) & ~(size_t)255);
  __bf16* w1b = (__bf16*)w;   w += (size_t)H_ * (2 * H_) * sizeof(__bf16);
  __bf16* w2b = (__bf16*)w;   w += (size_t)(2 * H_) * H_ * sizeof(__bf16);
  __bf16* owb = (__bf16*)w;   w += (size_t)H_ * NPADV * sizeof(__bf16);    // ~26 MB

  ms_swizzle_kernel<<<512, 256, 0, stream>>>(W1, w1b, H_, 2 * H_, 2 * H_);
  ms_swizzle_kernel<<<512, 256, 0, stream>>>(W2, w2b, 2 * H_, H_, H_);
  ms_swizzle_kernel<<<8192, 256, 0, stream>>>(out_w, owb, H_, V_, NPADV);

  ms_ffn_ln_kernel<<<(B_ * T_) / 32, 32, 0, stream>>>(seq, embed, w1b, b1, w2b, b2,
                                                      ln_g, ln_b, gate_w, gate_b,
                                                      hidden, scores);
  ms_topk_kernel<<<B_, 256, 0, stream>>>(scores, topidx);
  ms_query_kernel<<<B_, 256, 0, stream>>>(hidden, qa_w, qa_b, qr_w, qr_b, qbuf);
  ms_attn_kernel<<<B_, 256, 0, stream>>>(hidden, topidx, qbuf, ctxb);
  ms_outproj_kernel<<<(NPADV / 16 + 3) / 4, 128, 0, stream>>>(ctxb, owb, out_b, out);

  (void)in_sizes; (void)n_in; (void)out_size; (void)ws_size;
}
